// Head_6631429505133
// MI455X (gfx1250) — matrix operand order repaired and checked by reading, not compiled, not run on hardware
//
#include <hip/hip_runtime.h>
#include <hip/hip_bf16.h>

// Problem constants (reference: B=256, T=256, C=512, H=64, fp32 in/out)
#define BB 256
#define TT 256
#define CC 512
#define HH 64

typedef __bf16 bf16_t;
typedef __attribute__((ext_vector_type(16))) __bf16 v16bf;
typedef __attribute__((ext_vector_type(8)))  __bf16 bf16x8;
typedef __attribute__((ext_vector_type(8)))  float  v8f;
typedef int v4i __attribute__((vector_size(16)));   // matches builtin param type

#define AS1 __attribute__((address_space(1)))
#define AS3 __attribute__((address_space(3)))

#if defined(__has_builtin)
#if __has_builtin(__builtin_amdgcn_global_load_async_to_lds_b128)
#define HAVE_ASYNC_LDS 1
#endif
#endif
#ifndef HAVE_ASYNC_LDS
#define HAVE_ASYNC_LDS 0
#endif

// 16-byte global -> LDS copy: async DMA (ASYNCcnt) when available, else
// regular load + ds_store (still shares the tile across all 4 waves).
__device__ __forceinline__ void copy16_g2l(const bf16_t* g, bf16_t* l) {
#if HAVE_ASYNC_LDS
    __builtin_amdgcn_global_load_async_to_lds_b128(
        (AS1 v4i*)(unsigned long long)(uintptr_t)g,
        (AS3 v4i*)(unsigned int)(uintptr_t)l, 0, 0);
#else
    *(bf16x8*)l = *(const bf16x8*)g;
#endif
}

__device__ __forceinline__ void wait_async0() {
#if HAVE_ASYNC_LDS
#if __has_builtin(__builtin_amdgcn_s_wait_asynccnt)
    __builtin_amdgcn_s_wait_asynccnt(0);
#else
    asm volatile("s_wait_asynccnt 0" ::: "memory");
#endif
#endif
}

// Concatenate two aligned 8x bf16 runs into a 16-element WMMA fragment.
__device__ __forceinline__ v16bf frag_cat(const bf16_t* p0, const bf16_t* p1) {
    bf16x8 a = *(const bf16x8*)p0;
    bf16x8 b = *(const bf16x8*)p1;
    v16bf r;
#pragma unroll
    for (int i = 0; i < 8; ++i) { r[i] = a[i]; r[i + 8] = b[i]; }
    return r;
}

// ---------------------------------------------------------------------------
// Kernel 1: QKV projection.  Y_p = x @ W_p  (M=B*T=65536, K=C=512, N=H=64)
// 8 waves / block; weights staged transposed in 192 KB of CDNA5's 320 KB LDS.
// Q,K stored row-major bf16 [B*T,H]; V stored transposed bf16 [B,H,T].
// ---------------------------------------------------------------------------
__global__ void __launch_bounds__(256) qkv_proj_kernel(
    const float* __restrict__ x,
    const float* __restrict__ Wq, const float* __restrict__ Wk,
    const float* __restrict__ Wv,
    bf16_t* __restrict__ qout, bf16_t* __restrict__ kout,
    bf16_t* __restrict__ vtout)
{
    extern __shared__ bf16_t wt[];          // [3][HH][CC] transposed weights
    const int tid = threadIdx.x;
#pragma unroll
    for (int p = 0; p < 3; ++p) {
        const float* W = (p == 0) ? Wq : (p == 1) ? Wk : Wv;
        for (int idx = tid; idx < CC * HH; idx += 256) {
            const int kk = idx >> 6;        // row of W  (K index)
            const int n  = idx & 63;        // col of W  (N index)
            wt[p * (HH * CC) + n * CC + kk] = (bf16_t)W[idx];
        }
    }
    __syncthreads();

    const int wave = tid >> 5;
    const int lane = tid & 31;
    const int ln   = lane & 15;
    const int hi   = lane >> 4;
    const int m0   = blockIdx.x * 128 + wave * 16;   // global row base

    const v8f vzero = {0.f, 0.f, 0.f, 0.f, 0.f, 0.f, 0.f, 0.f};
    v8f acc[3][4];
#pragma unroll
    for (int p = 0; p < 3; ++p)
#pragma unroll
        for (int nt = 0; nt < 4; ++nt) acc[p][nt] = vzero;

    const float* xrow = x + (size_t)(m0 + ln) * CC;
    for (int k0 = 0; k0 < CC; k0 += 32) {
        // A fragment: lane = row, K pattern {kb..kb+7, 16+kb..16+kb+7}, kb=8*hi
        v16bf a;
        {
            const float* xp = xrow + k0 + hi * 8;
            float4 x0 = *(const float4*)(xp);
            float4 x1 = *(const float4*)(xp + 4);
            float4 x2 = *(const float4*)(xp + 16);
            float4 x3 = *(const float4*)(xp + 20);
            a[0]=(bf16_t)x0.x;  a[1]=(bf16_t)x0.y;  a[2]=(bf16_t)x0.z;  a[3]=(bf16_t)x0.w;
            a[4]=(bf16_t)x1.x;  a[5]=(bf16_t)x1.y;  a[6]=(bf16_t)x1.z;  a[7]=(bf16_t)x1.w;
            a[8]=(bf16_t)x2.x;  a[9]=(bf16_t)x2.y;  a[10]=(bf16_t)x2.z; a[11]=(bf16_t)x2.w;
            a[12]=(bf16_t)x3.x; a[13]=(bf16_t)x3.y; a[14]=(bf16_t)x3.z; a[15]=(bf16_t)x3.w;
        }
#pragma unroll
        for (int p = 0; p < 3; ++p) {
#pragma unroll
            for (int nt = 0; nt < 4; ++nt) {
                // B fragment: lane = column, K pattern {kb..kb+7, 16+kb..16+kb+7}, kb=8*hi
                const bf16_t* wp = wt + p * (HH * CC) + (nt * 16 + ln) * CC
                                   + k0 + hi * 8;
                v16bf bf = frag_cat(wp, wp + 16);
                acc[p][nt] = __builtin_amdgcn_wmma_f32_16x16x32_bf16(
                    false, a, false, bf, (short)0, acc[p][nt], false, false);
            }
        }
    }

    // C/D layout: lane = column (ln), VGPR r = row r + 8*hi
    const int bidx = m0 >> 8;                   // batch
    const int t0   = (m0 & (TT - 1)) + hi * 8;  // seq pos base for this lane
#pragma unroll
    for (int nt = 0; nt < 4; ++nt) {
        const int col = nt * 16 + ln;
#pragma unroll
        for (int r = 0; r < 8; ++r) {
            const int row = m0 + hi * 8 + r;
            qout[(size_t)row * HH + col] = (bf16_t)acc[0][nt][r];
            kout[(size_t)row * HH + col] = (bf16_t)acc[1][nt][r];
        }
        bf16x8 vp;
#pragma unroll
        for (int r = 0; r < 8; ++r) vp[r] = (bf16_t)acc[2][nt][r];
        *(bf16x8*)(vtout + ((size_t)bidx * HH + col) * TT + t0) = vp;
    }
}

// ---------------------------------------------------------------------------
// Kernel 2: causal flash attention.  Block = 4 waves = one 64-query tile of
// one batch.  K/V tiles are staged once per block into double-buffered LDS
// via async global->LDS DMA (ASYNCcnt) so the next tile's copy overlaps the
// current tile's WMMAs + online softmax.  P transposed to A-layout through
// wave-private LDS.  All GEMMs on bf16 WMMA, f32 accumulation.
// ---------------------------------------------------------------------------
__global__ void __launch_bounds__(128) attn_kernel(
    const bf16_t* __restrict__ qb, const bf16_t* __restrict__ kb,
    const bf16_t* __restrict__ vt, float* __restrict__ out)
{
    __shared__ bf16_t kt[2][64 * 64];     // K tile   [key][h]   (8 KB x2)
    __shared__ bf16_t vl[2][64 * 64];     // V^T tile [h][key]   (8 KB x2)
    __shared__ bf16_t plds[4 * 16 * 64];  // P tiles, wave-private (8 KB)

    const int tid  = threadIdx.x;
    const int wave = tid >> 5, lane = tid & 31, ln = lane & 15, hi = lane >> 4;
    const int b       = blockIdx.y;
    const int qt      = blockIdx.x;            // query tile (64 rows) index
    const int rowBase = qt * 64 + wave * 16;   // this wave's first query row
    const size_t qkBase = (size_t)b * TT * HH;
    const size_t vtBase = (size_t)b * HH * TT;
    const float scale = 0.044194173824159216f; // C^-0.5 (reference scaling)

    // Q A-fragments for this wave's 16 rows, two K-chunks over H=64
    v16bf qf[2];
    {
        const bf16_t* qrow = qb + qkBase + (size_t)(rowBase + ln) * HH;
#pragma unroll
        for (int c = 0; c < 2; ++c)
            qf[c] = frag_cat(qrow + c * 32 + hi * 8, qrow + c * 32 + hi * 8 + 16);
    }

    const v8f vzero = {0.f, 0.f, 0.f, 0.f, 0.f, 0.f, 0.f, 0.f};
    v8f oacc[4];
#pragma unroll
    for (int nt = 0; nt < 4; ++nt) oacc[nt] = vzero;
    float mrow[8], lrow[8];
#pragma unroll
    for (int r = 0; r < 8; ++r) { mrow[r] = -3.0e38f; lrow[r] = 0.f; }

    bf16_t* pl = plds + wave * (16 * 64);      // wave-private P tile [16][64]

    // Cooperative tile copy: K tile is one contiguous 8 KB block (full rows),
    // V^T tile is 64 rows of 128 B.  128 threads x 4 chunks x 16 B each.
    auto copy_tile = [&](int jb, int bufi) {
#pragma unroll
        for (int i = 0; i < 4; ++i) {
            const int c16 = i * 128 + tid;                 // 0..511
            copy16_g2l(kb + qkBase + (size_t)jb * HH + c16 * 8,
                       &kt[bufi][c16 * 8]);
            const int h = c16 >> 3, off = (c16 & 7) * 8;
            copy16_g2l(vt + vtBase + (size_t)h * TT + jb + off,
                       &vl[bufi][h * 64 + off]);
        }
    };

    copy_tile(0, 0);                           // prologue: tile 0 -> buf 0
    int buf = 0;

    for (int j = 0; j <= qt; ++j) {
        const int jb = j * 64;
        wait_async0();                         // this wave's DMA into buf done
        __syncthreads();                       // all waves' DMA done, prev reads done
        if (j < qt) copy_tile(jb + 64, buf ^ 1);   // overlap next tile's DMA

        // ---- S = Q @ K^T  (64 keys per tile) ----
        v8f s[4];
#pragma unroll
        for (int nt = 0; nt < 4; ++nt) s[nt] = vzero;
#pragma unroll
        for (int c = 0; c < 2; ++c) {
#pragma unroll
            for (int nt = 0; nt < 4; ++nt) {
                const bf16_t* kp = &kt[buf][(nt * 16 + ln) * 64 + c * 32 + hi * 8];
                v16bf kf = frag_cat(kp, kp + 16);
                s[nt] = __builtin_amdgcn_wmma_f32_16x16x32_bf16(
                    false, qf[c], false, kf, (short)0, s[nt], false, false);
            }
        }

        // ---- scale + causal mask (diagonal tile only) ----
        const bool diag = (j == qt);
#pragma unroll
        for (int nt = 0; nt < 4; ++nt) {
            const int col = jb + nt * 16 + ln;
#pragma unroll
            for (int r = 0; r < 8; ++r) {
                float v = s[nt][r] * scale;
                if (diag && col > rowBase + r + 8 * hi) v = -3.0e38f;
                s[nt][r] = v;
            }
        }

        // ---- online softmax per row (rows live across 16-lane halves) ----
#pragma unroll
        for (int r = 0; r < 8; ++r) {
            float mx = fmaxf(fmaxf(s[0][r], s[1][r]), fmaxf(s[2][r], s[3][r]));
#pragma unroll
            for (int off = 8; off >= 1; off >>= 1)
                mx = fmaxf(mx, __shfl_xor(mx, off, 32));
            const float mnew = fmaxf(mrow[r], mx);
            const float corr = __expf(mrow[r] - mnew);
            float rs = 0.f;
#pragma unroll
            for (int nt = 0; nt < 4; ++nt) {
                const float e = __expf(s[nt][r] - mnew);
                s[nt][r] = e;
                rs += e;
            }
#pragma unroll
            for (int off = 8; off >= 1; off >>= 1)
                rs += __shfl_xor(rs, off, 32);
            lrow[r] = lrow[r] * corr + rs;
            mrow[r] = mnew;
#pragma unroll
            for (int nt = 0; nt < 4; ++nt) oacc[nt][r] *= corr;
        }

        // ---- P (bf16) -> wave-private LDS (C/D layout -> row major) ----
#pragma unroll
        for (int nt = 0; nt < 4; ++nt)
#pragma unroll
            for (int r = 0; r < 8; ++r)
                pl[(r + 8 * hi) * 64 + nt * 16 + ln] = (bf16_t)s[nt][r];

        // ---- O += P @ V  (V^T tile in LDS -> contiguous B fragments) ----
#pragma unroll
        for (int c = 0; c < 2; ++c) {
            const bf16_t* pp = pl + ln * 64 + c * 32 + hi * 8;   // A-layout read
            v16bf pf = frag_cat(pp, pp + 16);
#pragma unroll
            for (int nt = 0; nt < 4; ++nt) {
                const bf16_t* vp = &vl[buf][(nt * 16 + ln) * 64 + c * 32 + hi * 8];
                v16bf vf = frag_cat(vp, vp + 16);
                oacc[nt] = __builtin_amdgcn_wmma_f32_16x16x32_bf16(
                    false, pf, false, vf, (short)0, oacc[nt], false, false);
            }
        }
        buf ^= 1;
    }

    // ---- normalize and write out (f32, [B,T,H]) ----
#pragma unroll
    for (int r = 0; r < 8; ++r) {
        const float inv = 1.0f / lrow[r];
        const int row = rowBase + r + 8 * hi;
#pragma unroll
        for (int nt = 0; nt < 4; ++nt)
            out[qkBase + (size_t)row * HH + nt * 16 + ln] = oacc[nt][r] * inv;
    }
}

// ---------------------------------------------------------------------------
extern "C" void kernel_launch(void* const* d_in, const int* in_sizes, int n_in,
                              void* d_out, int out_size, void* d_ws, size_t ws_size,
                              hipStream_t stream) {
    const float* x  = (const float*)d_in[0];
    const float* Wk = (const float*)d_in[1];
    const float* Wq = (const float*)d_in[2];
    const float* Wv = (const float*)d_in[3];
    float* out = (float*)d_out;

    // Workspace: bf16 Q [B*T,H], K [B*T,H], V^T [B,H,T]  (3 * 8 MB = 24 MB)
    bf16_t* qb = (bf16_t*)d_ws;
    bf16_t* kb = qb + (size_t)BB * TT * HH;
    bf16_t* vt = kb + (size_t)BB * TT * HH;

    const int wlds = 3 * CC * HH * (int)sizeof(bf16_t);   // 192 KB dynamic LDS
    (void)hipFuncSetAttribute(reinterpret_cast<const void*>(qkv_proj_kernel),
                              hipFuncAttributeMaxDynamicSharedMemorySize, wlds);

    qkv_proj_kernel<<<dim3((BB * TT) / 128), dim3(256), wlds, stream>>>(
        x, Wq, Wk, Wv, qb, kb, vt);
    attn_kernel<<<dim3(TT / 64, BB), dim3(128), 0, stream>>>(qb, kb, vt, out);
}
